// TransformerBlock_26560077758888
// MI455X (gfx1250) — matrix operand order repaired and checked by reading, not compiled, not run on hardware
//
#include <hip/hip_runtime.h>
#include <hip/hip_bf16.h>
#include <math.h>

// ---------------------------------------------------------------------------
// Types for WMMA operands (gfx1250 wave32: A/B = 16 f16 per lane, C/D = 8 f32)
// ---------------------------------------------------------------------------
typedef _Float16 h8v  __attribute__((ext_vector_type(8)));
typedef _Float16 h16v __attribute__((ext_vector_type(16)));
typedef float    f8v  __attribute__((ext_vector_type(8)));

#define WMMA_F16(A, B, C) \
  __builtin_amdgcn_wmma_f32_16x16x32_f16(false, (A), false, (B), (short)0, (C), false, false)

static __device__ __forceinline__ f8v zero8() {
  f8v z;
#pragma unroll
  for (int i = 0; i < 8; ++i) z[i] = 0.0f;
  return z;
}

// B-operand (32x16, K x N): lane holds column n = lane&15, elements 0..7 are
// K = half*8+0..7 and elements 8..15 are K = 16+half*8+0..7 (p = base+half*16).
static __device__ __forceinline__ h16v load_b16(const _Float16* p) {
  const int hb = (threadIdx.x >> 4) & 1;
  h8v lo = *(const h8v*)(p - 8 * hb);
  h8v hi = *(const h8v*)(p + 16 - 8 * hb);
  return __builtin_shufflevector(lo, hi, 0, 1, 2, 3, 4, 5, 6, 7,
                                 8, 9, 10, 11, 12, 13, 14, 15);
}

// A-operand (16x32, M x K): lane holds row m = lane&15, elements 0..7 are
// K = half*8+0..7 and elements 8..15 are K = 16+half*8+0..7 -> two 16B loads.
static __device__ __forceinline__ h16v load_a16(const _Float16* p0, const _Float16* p1) {
  h8v lo = *(const h8v*)p0;
  h8v hi = *(const h8v*)p1;
  return __builtin_shufflevector(lo, hi, 0, 1, 2, 3, 4, 5, 6, 7,
                                 8, 9, 10, 11, 12, 13, 14, 15);
}

// ---------------------------------------------------------------------------
// f32 -> f16 conversion (weights)
// ---------------------------------------------------------------------------
__global__ __launch_bounds__(256) void f32_to_f16_kernel(
    const float* __restrict__ in, _Float16* __restrict__ out, int n) {
  for (int i = blockIdx.x * blockDim.x + threadIdx.x; i < n;
       i += gridDim.x * blockDim.x)
    out[i] = (_Float16)in[i];
}

// ---------------------------------------------------------------------------
// LayerNorm over C=512 (no affine), f32 in -> f16 out. One block per row.
// ---------------------------------------------------------------------------
__global__ __launch_bounds__(256) void layernorm_f16_kernel(
    const float* __restrict__ x, _Float16* __restrict__ out) {
  const int row = blockIdx.x;
  const int t   = threadIdx.x;                    // 256 threads, 2 elems each
  const float* xr = x + (size_t)row * 512;
  float v0 = xr[t], v1 = xr[t + 256];

  __shared__ float red[8];
  const int wv = t >> 5, ln = t & 31;

  float s = v0 + v1;
#pragma unroll
  for (int m = 16; m >= 1; m >>= 1) s += __shfl_xor(s, m, 32);
  if (ln == 0) red[wv] = s;
  __syncthreads();
  float tot = 0.0f;
#pragma unroll
  for (int i = 0; i < 8; ++i) tot += red[i];
  const float mu = tot * (1.0f / 512.0f);
  __syncthreads();

  const float d0 = v0 - mu, d1 = v1 - mu;
  float q = d0 * d0 + d1 * d1;
#pragma unroll
  for (int m = 16; m >= 1; m >>= 1) q += __shfl_xor(q, m, 32);
  if (ln == 0) red[wv] = q;
  __syncthreads();
  float tot2 = 0.0f;
#pragma unroll
  for (int i = 0; i < 8; ++i) tot2 += red[i];
  const float rstd = rsqrtf(tot2 * (1.0f / 512.0f) + 1e-5f);

  out[(size_t)row * 512 + t]       = (_Float16)(d0 * rstd);
  out[(size_t)row * 512 + t + 256] = (_Float16)(d1 * rstd);
}

// ---------------------------------------------------------------------------
// Generic WMMA GEMM: out[M,N] = A[M,K](f16) @ W[N,K]^T(f16) (+bias)(+GELU)(+res)
// One wave computes a 16x64 tile (4 accumulators, A reused 4x).
// ---------------------------------------------------------------------------
template <bool HAS_BIAS, bool ADD_RES, bool GELU_ACT, bool OUT_F16>
__global__ __launch_bounds__(256) void gemm_wmma_kernel(
    const _Float16* __restrict__ A, const _Float16* __restrict__ W,
    const float* __restrict__ bias, const float* __restrict__ res,
    void* __restrict__ out, int M, int N, int K) {
  const int wid  = (blockIdx.x * blockDim.x + threadIdx.x) >> 5;
  const int lane = threadIdx.x & 31;
  const int idx  = lane & 15;
  const int hlf  = lane >> 4;
  const int tilesN = N >> 6;
  const int m0 = (wid / tilesN) << 4;
  const int n0 = (wid % tilesN) << 6;
  if (m0 >= M) return;

  f8v acc[4];
#pragma unroll
  for (int j = 0; j < 4; ++j) acc[j] = zero8();

  const _Float16* arow = A + (size_t)(m0 + idx) * K;
  for (int k0 = 0; k0 < K; k0 += 32) {
    if (k0 + 32 < K) __builtin_prefetch(arow + k0 + 32, 0, 0);
    const h16v a = load_a16(arow + k0 + hlf * 8, arow + k0 + 16 + hlf * 8);
    const _Float16* wb = W + (size_t)k0 + hlf * 16;
#pragma unroll
    for (int j = 0; j < 4; ++j) {
      const h16v b = load_b16(wb + (size_t)(n0 + j * 16 + idx) * K);
      acc[j] = WMMA_F16(a, b, acc[j]);
    }
  }

#pragma unroll
  for (int j = 0; j < 4; ++j) {
    const int n = n0 + j * 16 + idx;
    const float bv = HAS_BIAS ? bias[n] : 0.0f;
#pragma unroll
    for (int r = 0; r < 8; ++r) {
      const int mrow = m0 + r + 8 * hlf;
      float v = acc[j][r] + bv;
      if (GELU_ACT) v = 0.5f * v * (1.0f + erff(v * 0.70710678118654752f));
      if (ADD_RES) v += res[(size_t)mrow * N + n];
      if (OUT_F16)
        ((_Float16*)out)[(size_t)mrow * N + n] = (_Float16)v;
      else
        ((float*)out)[(size_t)mrow * N + n] = v;
    }
  }
}

// ---------------------------------------------------------------------------
// kqv [B*T, 3*512] f16  ->  K,Q row-major [BH,T,64] and V^T [BH,64,T].
// One block per (bh, 64-token tile); V transposed through padded LDS so all
// global reads/writes are 64B-contiguous per wave.
// ---------------------------------------------------------------------------
__global__ __launch_bounds__(256) void scatter_kqv_kernel(
    const _Float16* __restrict__ kqv, _Float16* __restrict__ k16,
    _Float16* __restrict__ q16, _Float16* __restrict__ vT16) {
  const int T = 2048;
  __shared__ _Float16 tile[64 * 65];          // 64x64 tile, pad to kill conflicts

  const int blk = blockIdx.x;                 // BH * (T/64) = 32*32 = 1024
  const int bh  = blk >> 5;
  const int t0  = (blk & 31) << 6;
  const int b   = bh >> 3, h = bh & 7;
  const int c   = threadIdx.x & 63;           // fast-moving column
  const int r4  = threadIdx.x >> 6;           // 0..3

#pragma unroll
  for (int i = 0; i < 16; ++i) {
    const int tt  = i * 4 + r4;
    const int tok = b * T + t0 + tt;
    const _Float16* src = kqv + (size_t)tok * 1536 + h * 64 + c;
    k16[((size_t)bh * T + t0 + tt) * 64 + c] = src[0];
    q16[((size_t)bh * T + t0 + tt) * 64 + c] = src[512];
    tile[tt * 65 + c] = src[1024];
  }
  __syncthreads();
#pragma unroll
  for (int i = 0; i < 16; ++i) {
    const int dd = i * 4 + r4;
    vT16[((size_t)bh * 64 + dd) * T + t0 + c] = tile[c * 65 + dd];
  }
}

// ---------------------------------------------------------------------------
// Flash attention. One wave = one (b,h, 16-query tile).
// scores^T = K_tile @ Q^T so the softmaxed P lands element-wise in A-layout.
// Score tiles are computed sequentially to keep VGPR pressure low (no spills).
// ---------------------------------------------------------------------------
__global__ __launch_bounds__(256) void attention_kernel(
    const _Float16* __restrict__ k16, const _Float16* __restrict__ q16,
    const _Float16* __restrict__ vT16, _Float16* __restrict__ att) {
  const int T = 2048, D = 64, C = 512;
  const int wid  = (blockIdx.x * blockDim.x + threadIdx.x) >> 5;
  const int lane = threadIdx.x & 31;
  const int idx  = lane & 15;
  const int hlf  = lane >> 4;
  const int bh = wid / (T / 16);          // 0..31
  const int q0 = (wid % (T / 16)) * 16;

  // Q as B-operand (fixed for the whole K loop): lane = query q0+idx.
  const _Float16* qrow = q16 + ((size_t)bh * T + q0 + idx) * D;
  const h16v bq0 = load_b16(qrow + hlf * 16);        // d = 0..31
  const h16v bq1 = load_b16(qrow + 32 + hlf * 16);   // d = 32..63

  f8v acc[4];
#pragma unroll
  for (int dt = 0; dt < 4; ++dt) acc[dt] = zero8();
  float m = -INFINITY, l = 0.0f;

  for (int kt = 0; kt < T; kt += 32) {
    const _Float16* kr0 = k16 + ((size_t)bh * T + kt + idx) * D;

    // scores^T tile 0: keys kt..kt+15 (one transient A register set at a time)
    f8v s0 = zero8();
    {
      h16v ak = load_a16(kr0 + hlf * 8, kr0 + 16 + hlf * 8);
      s0 = WMMA_F16(ak, bq0, s0);
      ak = load_a16(kr0 + 32 + hlf * 8, kr0 + 48 + hlf * 8);
      s0 = WMMA_F16(ak, bq1, s0);
    }
    // scores^T tile 1: keys kt+16..kt+31
    f8v s1 = zero8();
    {
      const _Float16* kr1 = kr0 + 16 * D;
      h16v ak = load_a16(kr1 + hlf * 8, kr1 + 16 + hlf * 8);
      s1 = WMMA_F16(ak, bq0, s1);
      ak = load_a16(kr1 + 32 + hlf * 8, kr1 + 48 + hlf * 8);
      s1 = WMMA_F16(ak, bq1, s1);
    }

    // lane holds query = idx; keys {hlf*8+r} U {16+hlf*8+r}
    float tm = -INFINITY;
#pragma unroll
    for (int r = 0; r < 8; ++r) {
      s0[r] *= 0.125f;                       // 1/sqrt(64)
      s1[r] *= 0.125f;
      tm = fmaxf(tm, fmaxf(s0[r], s1[r]));
    }
    tm = fmaxf(tm, __shfl_xor(tm, 16, 32));
    const float mnew  = fmaxf(m, tm);
    const float alpha = __expf(m - mnew);

    h16v ap;                                 // P lands element-wise in A-layout
    float rs = 0.0f;
#pragma unroll
    for (int r = 0; r < 8; ++r) {
      const float e0 = __expf(s0[r] - mnew);
      const float e1 = __expf(s1[r] - mnew);
      rs += e0 + e1;
      ap[r]     = (_Float16)e0;
      ap[8 + r] = (_Float16)e1;
    }
    rs += __shfl_xor(rs, 16, 32);
    l = l * alpha + rs;
    m = mnew;

    // Rescale accumulators: acc[dt][r] belongs to query r + 8*hlf.
    float af[8];
#pragma unroll
    for (int r = 0; r < 8; ++r) af[r] = __shfl(alpha, 8 * hlf + r, 32);
#pragma unroll
    for (int dt = 0; dt < 4; ++dt)
#pragma unroll
      for (int r = 0; r < 8; ++r) acc[dt][r] *= af[r];

    // acc += P @ V : B-operand from V^T rows (contiguous keys).
#pragma unroll
    for (int dt = 0; dt < 4; ++dt) {
      const _Float16* vrow =
          vT16 + ((size_t)bh * D + dt * 16 + idx) * T + kt + hlf * 16;
      acc[dt] = WMMA_F16(ap, load_b16(vrow), acc[dt]);
    }
  }

  float lf[8];
#pragma unroll
  for (int r = 0; r < 8; ++r) lf[r] = __shfl(l, 8 * hlf + r, 32);

  const int b = bh >> 3, h = bh & 7;
#pragma unroll
  for (int dt = 0; dt < 4; ++dt)
#pragma unroll
    for (int r = 0; r < 8; ++r) {
      const int tok = b * T + q0 + r + 8 * hlf;
      att[(size_t)tok * C + h * D + dt * 16 + idx] =
          (_Float16)(acc[dt][r] / lf[r]);
    }
}

// ---------------------------------------------------------------------------
// Launch
// ---------------------------------------------------------------------------
extern "C" void kernel_launch(void* const* d_in, const int* in_sizes, int n_in,
                              void* d_out, int out_size, void* d_ws, size_t ws_size,
                              hipStream_t stream) {
  (void)in_sizes; (void)n_in; (void)out_size; (void)ws_size;
  const float* x       = (const float*)d_in[0];
  const float* w_kqv   = (const float*)d_in[1];
  const float* b_kqv   = (const float*)d_in[2];
  const float* w_proj  = (const float*)d_in[3];
  const float* b_proj  = (const float*)d_in[4];
  const float* w_fc    = (const float*)d_in[5];
  const float* b_fc    = (const float*)d_in[6];
  const float* w_cproj = (const float*)d_in[7];

  const int B = 4, T = 2048, C = 512, R = B * T;   // R = 8192 rows

  char* ws = (char*)d_ws;
  size_t off = 0;
  auto alloc = [&](size_t bytes) -> char* {
    char* p = ws + off;
    off += (bytes + 255) & ~(size_t)255;
    return p;
  };
  _Float16* h16  = (_Float16*)alloc((size_t)R * C * 2);       // LN output (reused)
  _Float16* wk16 = (_Float16*)alloc((size_t)3 * C * C * 2);
  _Float16* wp16 = (_Float16*)alloc((size_t)C * C * 2);
  _Float16* wf16 = (_Float16*)alloc((size_t)2 * C * C * 2);
  _Float16* wc16 = (_Float16*)alloc((size_t)2 * C * C * 2);
  char* bufB = alloc((size_t)R * 3 * C * 2);                  // kqv16 -> att16+fc16
  char* bufC = alloc((size_t)R * 3 * C * 2);                  // k/q/vT -> xmid

  _Float16* kqv16 = (_Float16*)bufB;
  _Float16* att16 = (_Float16*)bufB;                          // reuse after scatter
  _Float16* fc16  = (_Float16*)(bufB + (size_t)R * C * 2);
  _Float16* k16   = (_Float16*)bufC;
  _Float16* q16   = (_Float16*)(bufC + (size_t)R * C * 2);
  _Float16* vT16  = (_Float16*)(bufC + (size_t)2 * R * C * 2);
  float*    xmid  = (float*)bufC;                             // reuse after attention

  // 1) weights -> f16
  f32_to_f16_kernel<<<512, 256, 0, stream>>>(w_kqv,   wk16, 3 * C * C);
  f32_to_f16_kernel<<<512, 256, 0, stream>>>(w_proj,  wp16, C * C);
  f32_to_f16_kernel<<<512, 256, 0, stream>>>(w_fc,    wf16, 2 * C * C);
  f32_to_f16_kernel<<<512, 256, 0, stream>>>(w_cproj, wc16, 2 * C * C);

  // 2) ln1
  layernorm_f16_kernel<<<R, 256, 0, stream>>>(x, h16);

  // 3) kqv = ln1 @ w_kqv^T + b_kqv           (waves = 512*24, blocks = 1536)
  gemm_wmma_kernel<true, false, false, true><<<1536, 256, 0, stream>>>(
      h16, wk16, b_kqv, nullptr, kqv16, R, 3 * C, C);

  // 4) split into K,Q,V^T per-head layouts (LDS tile transpose for V)
  scatter_kqv_kernel<<<1024, 256, 0, stream>>>(kqv16, k16, q16, vT16);

  // 5) flash attention                        (waves = 4096, blocks = 512)
  attention_kernel<<<512, 256, 0, stream>>>(k16, q16, vT16, att16);

  // 6) xmid = x + att @ w_proj^T + b_proj     (blocks = 512)
  gemm_wmma_kernel<true, true, false, false><<<512, 256, 0, stream>>>(
      att16, wp16, b_proj, x, xmid, R, C, C);

  // 7) ln2
  layernorm_f16_kernel<<<R, 256, 0, stream>>>(xmid, h16);

  // 8) fc = gelu(ln2 @ w_fc^T + b_fc)         (blocks = 1024)
  gemm_wmma_kernel<true, false, true, true><<<1024, 256, 0, stream>>>(
      h16, wf16, b_fc, nullptr, fc16, R, 2 * C, C);

  // 9) out = xmid + fc @ w_cproj^T            (blocks = 512)
  gemm_wmma_kernel<false, true, false, false><<<512, 256, 0, stream>>>(
      fc16, wc16, nullptr, xmid, (float*)d_out, R, C, 2 * C);
}
